// XLMRCorefModel_9543417331985
// MI455X (gfx1250) — matrix operand order repaired and checked by reading, not compiled, not run on hardware
//
#include <hip/hip_runtime.h>
#include <hip/hip_bf16.h>

// Problem constants (fixed by setup_inputs)
#define B_    2
#define S_    256
#define H_    768
#define L_    50
#define E_    512
#define W_    10
#define NSPAN 2515            // number of spans per batch
#define MROWS (B_ * NSPAN)    // 5030 GEMM rows
#define MPAD  5056            // rows padded to multiple of 64
#define D3    2304            // 3*H
#define KSEL  102             // int(0.4 * S)
#define NCOLS (KSEL + 1)      // 103
#define EBLK  (D3 / 128)      // 18 e-tiles in the GEMM

typedef __attribute__((ext_vector_type(16))) __bf16 v16bf;
typedef __attribute__((ext_vector_type(8)))  float  v8f;
typedef __attribute__((ext_vector_type(4)))  unsigned v4u;
typedef __attribute__((ext_vector_type(8)))  unsigned v8u;

struct B32x2 { uint4 x, y; };   // 32 bytes == v16bf

__device__ __forceinline__ unsigned short f2bf(float f) {
    unsigned u = __builtin_bit_cast(unsigned, f);
    unsigned r = u + 0x7FFFu + ((u >> 16) & 1u);   // round-to-nearest-even
    return (unsigned short)(r >> 16);
}

// TDM 2D tile load: D# group0 (count=1, lds_addr, global_addr, type=2) +
// caller-supplied group1 (dims/strides/padding). 2-group form => <=2D tensor.
__device__ __forceinline__ void tdm_load_2d(unsigned lds_addr, const void* gaddr,
                                            v8u g1) {
    unsigned long long ga = (unsigned long long)(uintptr_t)gaddr;
    v4u g0;
    g0[0] = 1u;                                       // count=1, user mode
    g0[1] = lds_addr;                                 // LDS byte address
    g0[2] = (unsigned)ga;                             // global_addr[31:0]
    g0[3] = ((unsigned)(ga >> 32) & 0x01FFFFFFu) | 0x80000000u;  // [56:32] | type=2
    asm volatile("tensor_load_to_lds %0, %1" :: "s"(g0), "s"(g1) : "memory");
}

// ---------------------------------------------------------------- utilities
__global__ void zero_f32_kernel(float* p, int n) {
    int i = blockIdx.x * 256 + threadIdx.x;
    if (i < n) p[i] = 0.f;
}
__global__ void zero_u32_kernel(unsigned* p, long long n) {
    long long i = (long long)blockIdx.x * 256 + threadIdx.x;
    long long stride = (long long)gridDim.x * 256;
    for (; i < n; i += stride) p[i] = 0u;
}
__global__ void cvt_bf16_kernel(const float* __restrict__ src,
                                unsigned short* __restrict__ dst, int n) {
    for (int i = blockIdx.x * 256 + threadIdx.x; i < n; i += gridDim.x * 256)
        dst[i] = f2bf(src[i]);
}

// ------------------------------------------------------- span enumeration
__global__ void span_setup_kernel(int* starts, int* ends) {
    int i = threadIdx.x;
    if (i >= S_) return;
    int nw = min(W_, S_ - i);
    int base;
    if (i <= S_ - W_) base = i * W_;
    else { int r = S_ - i; base = NSPAN - r * (r + 1) / 2; }
    for (int w = 0; w < nw; ++w) { starts[base + w] = i; ends[base + w] = i + w; }
}

// --------------------------------------------------------------- GCN edges
__global__ __launch_bounds__(256) void gcn_kernel(
    const float* __restrict__ hidden, const int* __restrict__ esrc,
    const int* __restrict__ etgt, const int* __restrict__ elab,
    const float* __restrict__ gcn_W, const float* __restrict__ gcn_b,
    float* __restrict__ gcn_out)
{
    int e = blockIdx.x;                 // 0 .. B*E-1
    int b = e / E_;
    int src = esrc[e], tgt = etgt[e], lab = elab[e];
    if (lab < 0 || lab >= L_ || src < 0 || src >= S_ || tgt < 0 || tgt >= S_) return;

    __shared__ float sx[H_];
    int tid = threadIdx.x;
    for (int h = tid; h < H_; h += 256)
        sx[h] = hidden[((size_t)b * S_ + src) * H_ + h];
    __syncthreads();

    const float* Wl = gcn_W + (size_t)lab * H_ * H_;
    float* outp = gcn_out + ((size_t)b * S_ + tgt) * H_;
    for (int h = tid; h < H_; h += 256) {
        const float* wr = Wl + (size_t)h * H_;
        float acc = gcn_b[lab * H_ + h];
        for (int k = 0; k < H_; k += 4) {
            float4 w4 = *(const float4*)(wr + k);
            float4 x4 = *(const float4*)(sx + k);
            acc += w4.x * x4.x + w4.y * x4.y + w4.z * x4.z + w4.w * x4.w;
        }
        atomicAdd(outp + h, acc);
    }
}

// embeddings = hidden + relu(gcn_out)
__global__ void embed_kernel(const float* __restrict__ hidden,
                             const float* __restrict__ gcn_out,
                             float* __restrict__ emb) {
    int i = blockIdx.x * 256 + threadIdx.x;
    if (i < B_ * S_ * H_) emb[i] = hidden[i] + fmaxf(gcn_out[i], 0.f);
}

// Per-token attention logit: tscore[b,s] = emb[b,s]·attn_w + attn_b
__global__ __launch_bounds__(256) void tscore_kernel(
    const float* __restrict__ emb, const float* __restrict__ attn_w,
    const float* __restrict__ attn_b, float* __restrict__ tscore)
{
    int row = blockIdx.x;               // b*S + s
    __shared__ float red[256];
    int tid = threadIdx.x;
    const float* e = emb + (size_t)row * H_;
    float a = 0.f;
    for (int h = tid; h < H_; h += 256) a += e[h] * attn_w[h];
    red[tid] = a; __syncthreads();
    for (int s = 128; s > 0; s >>= 1) {
        if (tid < s) red[tid] += red[tid + s];
        __syncthreads();
    }
    if (tid == 0) tscore[row] = red[0] + attn_b[0];
}

// Span pooling: softmax over <=10 token logits; emit span_reps fp32 (row-major)
// and the bf16 copy TRANSPOSED: At[d][row] with row stride MPAD.
__global__ __launch_bounds__(256) void span_pool_kernel(
    const float* __restrict__ emb, const float* __restrict__ tscore,
    const int* __restrict__ starts, const int* __restrict__ ends,
    float* __restrict__ span_reps, unsigned short* __restrict__ At)
{
    int bn = blockIdx.x;                 // b*NSPAN + n
    int b = bn / NSPAN, n = bn % NSPAN;
    int st = starts[n], en = ends[n];
    int width = en - st + 1;
    __shared__ float sp[W_];
    int tid = threadIdx.x;

    float mx = -1e30f;
    for (int w = 0; w < width; ++w) mx = fmaxf(mx, tscore[b * S_ + st + w]);
    float sum = 0.f;
    for (int w = 0; w < width; ++w) sum += __expf(tscore[b * S_ + st + w] - mx);
    float inv = 1.f / sum;
    if (tid < width) sp[tid] = __expf(tscore[b * S_ + st + tid] - mx) * inv;
    __syncthreads();

    const float* eb = emb + (size_t)b * S_ * H_;
    float* outp = span_reps + (size_t)bn * D3;
    for (int h = tid; h < H_; h += 256) {
        float vs = eb[(size_t)st * H_ + h];
        float ve = eb[(size_t)en * H_ + h];
        float pool = 0.f;
        for (int w = 0; w < width; ++w) pool += sp[w] * eb[(size_t)(st + w) * H_ + h];
        outp[h] = vs; outp[H_ + h] = ve; outp[2 * H_ + h] = pool;
        At[(size_t)h            * MPAD + bn] = f2bf(vs);
        At[(size_t)(H_ + h)     * MPAD + bn] = f2bf(ve);
        At[(size_t)(2 * H_ + h) * MPAD + bn] = f2bf(pool);
    }
}

// ------------------------------------------------- WMMA GEMM + fused mention
// D^T tile: D[e][m] = sum_k W1[e][k] * span_reps[m][k]
// Block tile 128e x 64m; 8 waves, each 32e x 32m (4 accumulators).
// Tiles streamed by the Tensor Data Mover (wave 0 issues 2 descriptors per
// k-step; TDM hardware padding reproduces LDS row strides 40/72 bf16).
// Double-buffered: tile i+1 DMAs while tile i feeds 4 WMMAs per wave.
// Fused epilogue: part[eblk][m] = sum_e relu(D + b1[e]) * w2[e].
__global__ __launch_bounds__(256) void gemm_mention_kernel(
    const unsigned short* __restrict__ Wb,   // [2304 e][2304 k] bf16
    const unsigned short* __restrict__ At,   // [2304 k][MPAD m] bf16
    const float* __restrict__ b1, const float* __restrict__ w2,
    float* __restrict__ part)
{
    __shared__ unsigned short sW[2][128 * 40];   // 128 e-rows x 32 k (stride 40)
    __shared__ unsigned short sM[2][32 * 72];    // 32 k-rows x 64 m (stride 72)
    __shared__ float swpart[4 * 64];             // per e-strip column partials

    const int tid  = threadIdx.x;
    const int m0   = blockIdx.x * 64;
    const int e0   = blockIdx.y * 128;
    const int lane = tid & 31;
    const int wave = tid >> 5;               // 8 waves
    const int we   = (wave >> 1) * 32;       // wave e offset: 0/32/64/96
    const int mn   = (wave & 1) * 32;        // wave m offset: 0/32
    const int hi   = lane >> 4;
    const int lr   = lane & 15;

    // D# group1 for the W tile: 32 elems/row (64B) + 16B pad -> stride 40 bf16.
    //   data_size=1 (2B): bit16; pad_enable: bit20; pad_interval=3 (16 DW=64B);
    //   pad_amount=3 (4 DW=16B); tensor_dim0=2304; tensor_dim1=2304;
    //   tile_dim0=32; tile_dim1=128; stride0=2304.
    v8u g1W;
    g1W[0] = (1u << 16) | (1u << 20) | (3u << 22) | (3u << 25);
    g1W[1] = (unsigned)D3 << 16;             // tensor_dim0[15:0]
    g1W[2] = (unsigned)D3 << 16;             // tensor_dim0[31:16]=0 | tensor_dim1[15:0]
    g1W[3] = 32u << 16;                      // tensor_dim1[31:16]=0 | tile_dim0
    g1W[4] = 128u;                           // tile_dim1 | tile_dim2=0
    g1W[5] = (unsigned)D3;                   // tensor_dim0_stride[31:0]
    g1W[6] = 0u; g1W[7] = 0u;
    // D# group1 for the M tile: 64 elems/row (128B) + 16B pad -> stride 72 bf16.
    //   pad_interval=4 (32 DW=128B); tile_dim0=64; tile_dim1=32; stride0=MPAD.
    v8u g1M;
    g1M[0] = (1u << 16) | (1u << 20) | (4u << 22) | (3u << 25);
    g1M[1] = (unsigned)MPAD << 16;
    g1M[2] = (unsigned)D3 << 16;
    g1M[3] = 64u << 16;
    g1M[4] = 32u;
    g1M[5] = (unsigned)MPAD;
    g1M[6] = 0u; g1M[7] = 0u;

    unsigned ldsWb[2], ldsMb[2];
    #pragma unroll
    for (int p = 0; p < 2; ++p) {
        ldsWb[p] = (unsigned)(uintptr_t)(&sW[p][0]);
        ldsMb[p] = (unsigned)(uintptr_t)(&sM[p][0]);
    }

    v8f acc00 = {}, acc01 = {}, acc10 = {}, acc11 = {};

    auto issue_tile = [&](int buf, int k0) {
        tdm_load_2d(ldsWb[buf], Wb + (size_t)e0 * D3 + k0, g1W);
        tdm_load_2d(ldsMb[buf], At + (size_t)k0 * MPAD + m0, g1M);
    };

    if (wave == 0) issue_tile(0, 0);         // prologue: tile 0 -> buffer 0
    int cur = 0;
    const int NIT = D3 / 32;                 // 72 k-steps

    for (int it = 0; it < NIT; ++it) {
        __syncthreads();                     // buf (1-cur) no longer being read
        if (wave == 0) {
            if (it + 1 < NIT) {
                issue_tile(1 - cur, (it + 1) * 32);
                // TDM ops are in-order per wave: <=2 pending => tile `it` done
                asm volatile("s_wait_tensorcnt 0x2" ::: "memory");
            } else {
                asm volatile("s_wait_tensorcnt 0x0" ::: "memory");
            }
        }
        __syncthreads();                     // everyone sees tile `it`

        const unsigned short* sWb = &sW[cur][0];
        const unsigned short* sMb = &sM[cur][0];

        // A fragments: two contiguous 16B runs per lane
        const unsigned short* ap0 = sWb + (we + lr) * 40 + hi * 8;
        const unsigned short* ap1 = sWb + (we + 16 + lr) * 40 + hi * 8;
        B32x2 ta0; ta0.x = *(const uint4*)ap0; ta0.y = *(const uint4*)(ap0 + 16);
        B32x2 ta1; ta1.x = *(const uint4*)ap1; ta1.y = *(const uint4*)(ap1 + 16);
        v16bf af0 = __builtin_bit_cast(v16bf, ta0);
        v16bf af1 = __builtin_bit_cast(v16bf, ta1);
        // B fragments: column n = lr, k in table order (same as A)
        const unsigned short* bp = sMb + mn + lr;
        v16bf bf0, bf1;
        #pragma unroll
        for (int s = 0; s < 16; ++s) {
            int kk = (s < 8) ? (hi * 8 + s) : (16 + hi * 8 + (s - 8));
            bf0[s] = __builtin_bit_cast(__bf16, bp[kk * 72]);
            bf1[s] = __builtin_bit_cast(__bf16, bp[kk * 72 + 16]);
        }

        acc00 = __builtin_amdgcn_wmma_f32_16x16x32_bf16(false, af0, false, bf0,
                                                        (short)0, acc00, false, false);
        acc01 = __builtin_amdgcn_wmma_f32_16x16x32_bf16(false, af0, false, bf1,
                                                        (short)0, acc01, false, false);
        acc10 = __builtin_amdgcn_wmma_f32_16x16x32_bf16(false, af1, false, bf0,
                                                        (short)0, acc10, false, false);
        acc11 = __builtin_amdgcn_wmma_f32_16x16x32_bf16(false, af1, false, bf1,
                                                        (short)0, acc11, false, false);
        cur ^= 1;
    }

    // Fused epilogue: per-column sum over this block's 128 e of relu(c+b1)*w2.
    float cp0 = 0.f, cp1 = 0.f;
    #pragma unroll
    for (int es = 0; es < 2; ++es) {
        const v8f* a0 = es ? &acc10 : &acc00;
        const v8f* a1 = es ? &acc11 : &acc01;
        #pragma unroll
        for (int r = 0; r < 8; ++r) {
            int e = e0 + we + es * 16 + hi * 8 + r;
            float bb = b1[e], ww = w2[e];
            float h0 = (*a0)[r] + bb; h0 = h0 > 0.f ? h0 : 0.f;
            float h1 = (*a1)[r] + bb; h1 = h1 > 0.f ? h1 : 0.f;
            cp0 += h0 * ww; cp1 += h1 * ww;
        }
    }
    cp0 += __shfl_xor(cp0, 16, 32);          // fold hi/lo half-wave (same column)
    cp1 += __shfl_xor(cp1, 16, 32);
    if (hi == 0) {
        swpart[(wave >> 1) * 64 + mn + lr]      = cp0;
        swpart[(wave >> 1) * 64 + mn + 16 + lr] = cp1;
    }
    __syncthreads();
    if (tid < 64) {
        float s = swpart[tid] + swpart[64 + tid] + swpart[128 + tid] + swpart[192 + tid];
        part[(size_t)blockIdx.y * MPAD + m0 + tid] = s;
    }
}

__global__ void mention_reduce_kernel(const float* __restrict__ part,
                                      const float* __restrict__ ms_b2,
                                      float* __restrict__ mention) {
    int i = blockIdx.x * 256 + threadIdx.x;
    if (i >= MROWS) return;
    float s = ms_b2[0];
    for (int nb = 0; nb < EBLK; ++nb) s += part[(size_t)nb * MPAD + i];
    mention[i] = s;
}

// --------------------------------------------------------------- top-k (102)
__global__ __launch_bounds__(256) void topk_kernel(const float* __restrict__ mention,
                                                   int* __restrict__ top_idx) {
    __shared__ float vals[NSPAN];
    __shared__ float rv[256];
    __shared__ int   ri[256];
    int b = blockIdx.x, tid = threadIdx.x;
    for (int i = tid; i < NSPAN; i += 256) vals[i] = mention[b * NSPAN + i];
    __syncthreads();
    for (int t = 0; t < KSEL; ++t) {
        float bv = -INFINITY; int bi = NSPAN;
        for (int i = tid; i < NSPAN; i += 256) {
            float v = vals[i];
            if (v > bv || (v == bv && i < bi)) { bv = v; bi = i; }
        }
        rv[tid] = bv; ri[tid] = bi; __syncthreads();
        for (int s = 128; s > 0; s >>= 1) {
            if (tid < s) {
                float ov = rv[tid + s]; int oi = ri[tid + s];
                if (ov > rv[tid] || (ov == rv[tid] && oi < ri[tid])) { rv[tid] = ov; ri[tid] = oi; }
            }
            __syncthreads();
        }
        if (tid == 0) { top_idx[b * KSEL + t] = ri[0]; vals[ri[0]] = -INFINITY; }
        __syncthreads();
    }
}

// Antecedent scores: s_j = pruned·ant_w[:D], s_i = pruned·ant_w[D:]
__global__ __launch_bounds__(256) void antscore_kernel(
    const float* __restrict__ span_reps, const int* __restrict__ top_idx,
    const float* __restrict__ ant_w, float* __restrict__ sj, float* __restrict__ si)
{
    int bk = blockIdx.x;
    int b = bk / KSEL;
    int n = top_idx[bk];
    __shared__ float r0[256], r1[256];
    int tid = threadIdx.x;
    const float* row = span_reps + ((size_t)b * NSPAN + n) * D3;
    float a0 = 0.f, a1 = 0.f;
    for (int d = tid; d < D3; d += 256) {
        float r = row[d];
        a0 += r * ant_w[d];
        a1 += r * ant_w[D3 + d];
    }
    r0[tid] = a0; r1[tid] = a1; __syncthreads();
    for (int s = 128; s > 0; s >>= 1) {
        if (tid < s) { r0[tid] += r0[tid + s]; r1[tid] += r1[tid + s]; }
        __syncthreads();
    }
    if (tid == 0) { sj[bk] = r0[0]; si[bk] = r1[0]; }
}

// Masked log-softmax over 103 columns per (b,i)
__global__ __launch_bounds__(128) void final_kernel(
    const float* __restrict__ sj, const float* __restrict__ si,
    const float* __restrict__ ant_b, float* __restrict__ out)
{
    int bi = blockIdx.x;
    int b = bi / KSEL, i = bi % KSEL;
    int tid = threadIdx.x;
    __shared__ float sh[128];

    float myv = -INFINITY;
    if (tid < NCOLS) {
        if (tid == 0) myv = 0.f;                                  // dummy column
        else {
            int j = tid - 1;
            myv = (j >= i) ? -1e10f : (sj[b * KSEL + j] + si[b * KSEL + i] + ant_b[0]);
        }
    }
    sh[tid] = myv; __syncthreads();
    for (int s = 64; s > 0; s >>= 1) { if (tid < s) sh[tid] = fmaxf(sh[tid], sh[tid + s]); __syncthreads(); }
    float mx = sh[0]; __syncthreads();
    float e = (tid < NCOLS) ? __expf(myv - mx) : 0.f;
    sh[tid] = e; __syncthreads();
    for (int s = 64; s > 0; s >>= 1) { if (tid < s) sh[tid] += sh[tid + s]; __syncthreads(); }
    float denom = sh[0];
    if (tid < NCOLS) out[(size_t)bi * NCOLS + tid] = __logf(e / denom + 1e-10f);
}

// ------------------------------------------------------------------ launcher
extern "C" void kernel_launch(void* const* d_in, const int* in_sizes, int n_in,
                              void* d_out, int out_size, void* d_ws, size_t ws_size,
                              hipStream_t stream) {
    (void)in_sizes; (void)n_in; (void)out_size; (void)ws_size;
    const float* hidden = (const float*)d_in[0];
    const int*   esrc   = (const int*)d_in[2];
    const int*   etgt   = (const int*)d_in[3];
    const int*   elab   = (const int*)d_in[4];
    const float* gcn_W  = (const float*)d_in[5];
    const float* gcn_b  = (const float*)d_in[6];
    const float* attn_w = (const float*)d_in[7];
    const float* attn_b = (const float*)d_in[8];
    const float* ms_W1  = (const float*)d_in[9];
    const float* ms_b1  = (const float*)d_in[10];
    const float* ms_w2  = (const float*)d_in[11];
    const float* ms_b2  = (const float*)d_in[12];
    const float* ant_w  = (const float*)d_in[13];
    const float* ant_b  = (const float*)d_in[14];

    char* ws = (char*)d_ws;
    size_t off = 0;
    auto carve = [&](size_t bytes) -> void* {
        void* p = ws + off;
        off += (bytes + 255) & ~(size_t)255;
        return p;
    };
    float* gcn_out        = (float*)carve((size_t)B_ * S_ * H_ * 4);
    float* emb            = (float*)carve((size_t)B_ * S_ * H_ * 4);
    float* tscore         = (float*)carve((size_t)B_ * S_ * 4);
    int*   starts         = (int*)carve((size_t)NSPAN * 4);
    int*   ends           = (int*)carve((size_t)NSPAN * 4);
    float* span_reps      = (float*)carve((size_t)MROWS * D3 * 4);
    unsigned short* At_bf = (unsigned short*)carve((size_t)D3 * MPAD * 2);  // transposed
    unsigned short* W_bf  = (unsigned short*)carve((size_t)D3 * D3 * 2);
    float* part           = (float*)carve((size_t)EBLK * MPAD * 4);
    float* mention        = (float*)carve((size_t)MROWS * 4);
    int*   top_idx        = (int*)carve((size_t)B_ * KSEL * 4);
    float* sj             = (float*)carve((size_t)B_ * KSEL * 4);
    float* si             = (float*)carve((size_t)B_ * KSEL * 4);

    int nBSH = B_ * S_ * H_;
    zero_f32_kernel<<<(nBSH + 255) / 256, 256, 0, stream>>>(gcn_out, nBSH);
    zero_u32_kernel<<<4096, 256, 0, stream>>>((unsigned*)At_bf,
                                              (long long)D3 * MPAD / 2);
    span_setup_kernel<<<1, 256, 0, stream>>>(starts, ends);
    gcn_kernel<<<B_ * E_, 256, 0, stream>>>(hidden, esrc, etgt, elab, gcn_W, gcn_b, gcn_out);
    embed_kernel<<<(nBSH + 255) / 256, 256, 0, stream>>>(hidden, gcn_out, emb);
    tscore_kernel<<<B_ * S_, 256, 0, stream>>>(emb, attn_w, attn_b, tscore);
    cvt_bf16_kernel<<<2048, 256, 0, stream>>>(ms_W1, W_bf, D3 * D3);
    span_pool_kernel<<<MROWS, 256, 0, stream>>>(emb, tscore, starts, ends, span_reps, At_bf);
    dim3 ggrid(MPAD / 64, EBLK);
    gemm_mention_kernel<<<ggrid, 256, 0, stream>>>(W_bf, At_bf, ms_b1, ms_w2, part);
    mention_reduce_kernel<<<(MROWS + 255) / 256, 256, 0, stream>>>(part, ms_b2, mention);
    topk_kernel<<<B_, 256, 0, stream>>>(mention, top_idx);
    antscore_kernel<<<B_ * KSEL, 256, 0, stream>>>(span_reps, top_idx, ant_w, sj, si);
    final_kernel<<<B_ * KSEL, 128, 0, stream>>>(sj, si, ant_b, (float*)d_out);
}
